// MAB_7619271983203
// MI455X (gfx1250) — matrix operand order repaired and checked by reading, not compiled, not run on hardware
//
#include <hip/hip_runtime.h>
#include <hip/hip_bf16.h>

// ---------------------------------------------------------------------------
// MAB block for MI455X (gfx1250, wave32, WMMA).
// cast->bf16, WMMA GEMMs with double-buffered LDS weight staging via
// GLOBAL_LOAD_ASYNC_TO_LDS_B128 (ASYNCcnt), flash attention (WMMA + online
// softmax), fused residual LayerNorms.
// ---------------------------------------------------------------------------

typedef __bf16 bf16_t;
typedef __attribute__((ext_vector_type(16))) __bf16 v16bf;
typedef __attribute__((ext_vector_type(8)))  __bf16 v8bf;
typedef __attribute__((ext_vector_type(8)))  float  v8f;
typedef __attribute__((ext_vector_type(4)))  int    v4i;

static constexpr int kB  = 8;
static constexpr int kN  = 1024;
static constexpr int kM  = 1024;
static constexpr int kDQ = 512;
static constexpr int kDV = 512;
static constexpr int kH  = 8;
static constexpr int kDH = 64;     // head dim
static constexpr int kDFF = 2048;
static constexpr int kRows = kB * kN;   // 8192 token rows
static constexpr int kPad = 40;    // LDS row pitch in halves (80B: 16B-aligned,
                                   // m*20 mod 64 hits all banks -> conflict-free)

// ---------------- async global->LDS (probe-guarded) -----------------------
// Probe result (round 2 diagnostic): builtin exists, 4 args, arg0 is a
// non-const v4i pointer in the global address space.
#if __has_builtin(__builtin_amdgcn_global_load_async_to_lds_b128)
#define ATH_ASYNC_LDS 1
#else
#define ATH_ASYNC_LDS 0
#endif

__device__ __forceinline__ void stage16B(bf16_t* dst, const bf16_t* src) {
#if ATH_ASYNC_LDS
  __builtin_amdgcn_global_load_async_to_lds_b128(
      (__attribute__((address_space(1))) v4i*)src,
      (__attribute__((address_space(3))) v4i*)dst, 0, 0);
#else
  *(v8bf*)dst = *(const v8bf*)src;
#endif
}

__device__ __forceinline__ void wait_stage() {
#if ATH_ASYNC_LDS
#if __has_builtin(__builtin_amdgcn_s_wait_asynccnt)
  __builtin_amdgcn_s_wait_asynccnt(0);
#else
  asm volatile("s_wait_asynccnt 0x0" ::: "memory");
#endif
#endif
}

// ---------------- WMMA helpers -------------------------------------------
__device__ __forceinline__ v8f wmma_bf16(v16bf a, v16bf b, v8f c) {
  // (neg_a, A, neg_b, B, c_mod, C, reuse_a, reuse_b)
  return __builtin_amdgcn_wmma_f32_16x16x32_bf16(false, a, false, b,
                                                 (short)0, c, false, false);
}

__device__ __forceinline__ v16bf frag_pair(const bf16_t* p0, const bf16_t* p1) {
  union { v16bf f; v8bf h[2]; } u;
  u.h[0] = *(const v8bf*)p0;
  u.h[1] = *(const v8bf*)p1;
  return u.f;
}

// A-matrix 16x32 bf16 fragment. Lane (m = lane&15, hi = lane>>4) holds row m.
// halves 0..7 -> K = hi*8 + 0..7 ; halves 8..15 -> K = 16 + hi*8 + 0..7.
__device__ __forceinline__ v16bf load_fragA(const bf16_t* row, int hi) {
  return frag_pair(row + hi * 8, row + 16 + hi * 8);
}

// B-matrix 32x16 bf16 fragment: lane n holds column n; halves 0..7 -> K = hi*8 + 0..7 ; halves 8..15 -> K = 16 + hi*8 + 0..7.
__device__ __forceinline__ v16bf load_fragB(const bf16_t* row, int hi) {
  return frag_pair(row + hi * 8, row + 16 + hi * 8);
}

// ---------------- cast f32 -> bf16 ---------------------------------------
__global__ void cast_f32_bf16(const float* __restrict__ src,
                              bf16_t* __restrict__ dst, int n) {
  int i = blockIdx.x * blockDim.x + threadIdx.x;
  int stride = gridDim.x * blockDim.x;
  for (; i < n; i += stride) dst[i] = (bf16_t)src[i];
}

// ---------------- generic Linear GEMM: C = act(A @ W^T + bias) ------------
// A: [Mr, K] bf16 row-major. W: [N, K] bf16 row-major.
// Block = 8 waves = 128 rows x 64 cols; W tile (64x32) double-buffered in LDS
// (async global->LDS) and shared by all 8 waves.
template <bool RELU, bool WF32, bool WBF16>
__global__ __launch_bounds__(256)
void gemm_bias_act(const bf16_t* __restrict__ A, const bf16_t* __restrict__ W,
                   const float* __restrict__ bias,
                   float* __restrict__ Cf, bf16_t* __restrict__ Cb,
                   int N, int K) {
  __shared__ __align__(16) bf16_t Wt[2][64][kPad];

  const int wave = threadIdx.x >> 5;
  const int lane = threadIdx.x & 31;
  const int m  = lane & 15;
  const int hi = lane >> 4;
  const int row0 = blockIdx.y * 128 + wave * 16;
  const int col0 = blockIdx.x * 64;

  // staging role: thread t copies 8 halves (16B) of W tile
  const int scol  = threadIdx.x >> 2;        // 0..63
  const int skseg = (threadIdx.x & 3) * 8;   // 0,8,16,24
  const bf16_t* wsrc = W + (size_t)(col0 + scol) * K + skseg;

  const bf16_t* arow = A + (size_t)(row0 + m) * K;

  stage16B(&Wt[0][scol][skseg], wsrc);       // k = 0 tile

  v8f acc[4] = {v8f{}, v8f{}, v8f{}, v8f{}};
  for (int k = 0; k < K; k += 32) {
    const int buf = (k >> 5) & 1;
    wait_stage();
    __syncthreads();                         // tile `buf` visible to all waves
    if (k + 32 < K) stage16B(&Wt[buf ^ 1][scol][skseg], wsrc + k + 32);

    __builtin_prefetch(arow + k + 128, 0, 0);  // global_prefetch_b8
    v16bf af = load_fragA(arow + k, hi);
#pragma unroll
    for (int c = 0; c < 4; ++c) {
      v16bf bfrag = load_fragB(&Wt[buf][c * 16 + m][0], hi);
      acc[c] = wmma_bf16(af, bfrag, acc[c]);
    }
  }

#pragma unroll
  for (int c = 0; c < 4; ++c) {
    const int col = col0 + c * 16 + m;
    const float bv = bias[col];
#pragma unroll
    for (int r = 0; r < 8; ++r) {
      const int row = row0 + r + hi * 8;
      float v = acc[c][r] + bv;
      if (RELU) v = v > 0.0f ? v : 0.0f;
      const size_t idx = (size_t)row * N + col;
      if (WF32)  Cf[idx] = v;
      if (WBF16) Cb[idx] = (bf16_t)v;
    }
  }
}

// ---------------- flash attention ----------------------------------------
// grid = (N/64, H, B), block = 128 (4 waves). Each wave owns 16 query rows.
// Per 32-key tile: S = Q@K^T (4 WMMA), online softmax (shfl_xor row reduce
// over the 16-lane column groups), P via LDS relayout, P@V (4 WMMA).
__global__ __launch_bounds__(128)
void attention_kernel(const bf16_t* __restrict__ Qp, const bf16_t* __restrict__ Kp,
                      const bf16_t* __restrict__ Vp, float* __restrict__ O) {
  __shared__ __align__(16) bf16_t Vt[kDH][kPad];       // V tile transposed [hdim][key]
  __shared__ __align__(16) bf16_t Pl[4][16][kPad];     // per-wave P tile

  const int wave = threadIdx.x >> 5;
  const int lane = threadIdx.x & 31;
  const int m  = lane & 15;
  const int hi = lane >> 4;
  const int b = blockIdx.z, h = blockIdx.y;
  const int n0 = blockIdx.x * 64 + wave * 16;

  const bf16_t* qrow  = Qp + ((size_t)(b * kN + n0 + m)) * kDV + h * kDH;
  const bf16_t* kbase = Kp + ((size_t)(b * kM)) * kDV + h * kDH;
  const bf16_t* vbase = Vp + ((size_t)(b * kM)) * kDV + h * kDH;

  const v16bf qf0 = load_fragA(qrow, hi);        // head dims 0..31
  const v16bf qf1 = load_fragA(qrow + 32, hi);   // head dims 32..63

  float mrow[8], lrow[8];
  v8f acc[4] = {v8f{}, v8f{}, v8f{}, v8f{}};
#pragma unroll
  for (int r = 0; r < 8; ++r) { mrow[r] = -3.0e38f; lrow[r] = 0.0f; }

  const float scale = 0.125f;  // 1/sqrt(64)

  for (int kt = 0; kt < kM; kt += 32) {
    // ---- stage V tile transposed into LDS (cooperative, 128 threads) ----
    __syncthreads();   // previous-iteration Vt readers are done
    {
      const int key  = threadIdx.x >> 2;          // 0..31
      const int cseg = (threadIdx.x & 3) * 16;    // 0,16,32,48
      const bf16_t* vr = vbase + (size_t)(kt + key) * kDV + cseg;
      v8bf a0 = *(const v8bf*)(vr);
      v8bf a1 = *(const v8bf*)(vr + 8);
#pragma unroll
      for (int j = 0; j < 8; ++j) {
        Vt[cseg + j][key]     = a0[j];
        Vt[cseg + 8 + j][key] = a1[j];
      }
    }

    // ---- S = Q @ K^T for 2 key sub-blocks of 16 ----
    v8f s[2] = {v8f{}, v8f{}};
#pragma unroll
    for (int j = 0; j < 2; ++j) {
      const bf16_t* krow = kbase + (size_t)(kt + j * 16 + m) * kDV;
      s[j] = wmma_bf16(qf0, load_fragB(krow, hi), s[j]);
      s[j] = wmma_bf16(qf1, load_fragB(krow + 32, hi), s[j]);
    }
    s[0] = s[0] * scale;
    s[1] = s[1] * scale;

    __syncthreads();   // Vt staged before P@V below

    // ---- online softmax; rows live in element r (row = r + hi*8),
    //      columns across the 16-lane group -> shfl_xor reductions.
#pragma unroll
    for (int r = 0; r < 8; ++r) {
      float mt = fmaxf(s[0][r], s[1][r]);
#pragma unroll
      for (int off = 8; off >= 1; off >>= 1) mt = fmaxf(mt, __shfl_xor(mt, off));
      const float mn = fmaxf(mrow[r], mt);
      const float alpha = __expf(mrow[r] - mn);
      mrow[r] = mn;
      const float p0 = __expf(s[0][r] - mn);
      const float p1 = __expf(s[1][r] - mn);
      s[0][r] = p0;
      s[1][r] = p1;
      float psum = p0 + p1;
#pragma unroll
      for (int off = 8; off >= 1; off >>= 1) psum += __shfl_xor(psum, off);
      lrow[r] = lrow[r] * alpha + psum;
#pragma unroll
      for (int c = 0; c < 4; ++c) acc[c][r] *= alpha;
    }

    // ---- relayout P: C-form -> LDS -> A-form fragment ----
#pragma unroll
    for (int j = 0; j < 2; ++j)
#pragma unroll
      for (int r = 0; r < 8; ++r)
        Pl[wave][r + hi * 8][j * 16 + m] = (bf16_t)s[j][r];

    const v16bf pf = load_fragA(&Pl[wave][m][0], hi);   // 16x32 P fragment

    // ---- O += P @ V ----
#pragma unroll
    for (int c = 0; c < 4; ++c) {
      const v16bf vf = load_fragB(&Vt[c * 16 + m][0], hi);
      acc[c] = wmma_bf16(pf, vf, acc[c]);
    }
  }

  // ---- normalize and store ----
#pragma unroll
  for (int r = 0; r < 8; ++r) {
    const float inv = 1.0f / lrow[r];
    const int row = n0 + r + hi * 8;
#pragma unroll
    for (int c = 0; c < 4; ++c) {
      const int col = h * kDH + c * 16 + m;
      O[((size_t)(b * kN + row)) * kDV + col] = acc[c][r] * inv;
    }
  }
}

// ---------------- fused residual + LayerNorm ------------------------------
// One block (256 threads) per row of 512; each thread handles 2 elements.
template <bool WBF16>
__global__ __launch_bounds__(256)
void add_layernorm(const float* __restrict__ X, const float* __restrict__ R,
                   const float* __restrict__ g, const float* __restrict__ bta,
                   float* __restrict__ Yf, bf16_t* __restrict__ Yb) {
  const int row = blockIdx.x;
  const int t = threadIdx.x;
  const size_t base = (size_t)row * kDV;

  const float v0 = X[base + t] + R[base + t];
  const float v1 = X[base + t + 256] + R[base + t + 256];

  __shared__ float sred[8];
  __shared__ float sstat[2];

  float s = v0 + v1;
#pragma unroll
  for (int off = 16; off >= 1; off >>= 1) s += __shfl_xor(s, off);
  if ((t & 31) == 0) sred[t >> 5] = s;
  __syncthreads();
  if (t == 0) {
    float tot = 0.0f;
#pragma unroll
    for (int i = 0; i < 8; ++i) tot += sred[i];
    sstat[0] = tot * (1.0f / (float)kDV);
  }
  __syncthreads();
  const float mu = sstat[0];
  const float d0 = v0 - mu, d1 = v1 - mu;

  float q = d0 * d0 + d1 * d1;
#pragma unroll
  for (int off = 16; off >= 1; off >>= 1) q += __shfl_xor(q, off);
  if ((t & 31) == 0) sred[t >> 5] = q;
  __syncthreads();
  if (t == 0) {
    float tot = 0.0f;
#pragma unroll
    for (int i = 0; i < 8; ++i) tot += sred[i];
    sstat[1] = tot * (1.0f / (float)kDV);
  }
  __syncthreads();

  const float inv = rsqrtf(sstat[1] + 1e-5f);
  const float y0 = d0 * inv * g[t] + bta[t];
  const float y1 = d1 * inv * g[t + 256] + bta[t + 256];
  Yf[base + t] = y0;
  Yf[base + t + 256] = y1;
  if (WBF16) {
    Yb[base + t] = (bf16_t)y0;
    Yb[base + t + 256] = (bf16_t)y1;
  }
}

// ---------------- host driver ---------------------------------------------
extern "C" void kernel_launch(void* const* d_in, const int* in_sizes, int n_in,
                              void* d_out, int out_size, void* d_ws, size_t ws_size,
                              hipStream_t stream) {
  (void)in_sizes; (void)n_in; (void)out_size; (void)ws_size;

  const float* Q  = (const float*)d_in[0];
  const float* K  = (const float*)d_in[1];
  const float* Wq = (const float*)d_in[2];
  const float* bq = (const float*)d_in[3];
  const float* Wk = (const float*)d_in[4];
  const float* bk = (const float*)d_in[5];
  const float* Wv = (const float*)d_in[6];
  const float* bv = (const float*)d_in[7];
  const float* W1 = (const float*)d_in[8];
  const float* b1 = (const float*)d_in[9];
  const float* W2 = (const float*)d_in[10];
  const float* b2 = (const float*)d_in[11];
  const float* W3 = (const float*)d_in[12];
  const float* b3 = (const float*)d_in[13];
  const float* g0 = (const float*)d_in[14];
  const float* beta0 = (const float*)d_in[15];
  const float* g1 = (const float*)d_in[16];
  const float* beta1 = (const float*)d_in[17];
  float* out = (float*)d_out;

  // bump allocator over workspace (256B aligned)
  char* ws = (char*)d_ws;
  size_t off = 0;
  auto alloc = [&](size_t bytes) -> void* {
    void* p = ws + off;
    off += (bytes + 255) & ~(size_t)255;
    return p;
  };

  const size_t nTok = (size_t)kRows * kDV;        // 8192*512
  const size_t nFF  = (size_t)kRows * kDFF;       // 8192*2048

  bf16_t* Qbf  = (bf16_t*)alloc(nTok * 2);
  bf16_t* Kbf  = (bf16_t*)alloc(nTok * 2);
  bf16_t* Wqb  = (bf16_t*)alloc((size_t)kDV * kDQ * 2);
  bf16_t* Wkb  = (bf16_t*)alloc((size_t)kDV * kDQ * 2);
  bf16_t* Wvb  = (bf16_t*)alloc((size_t)kDV * kDQ * 2);
  bf16_t* W1b  = (bf16_t*)alloc((size_t)kDFF * kDV * 2);
  bf16_t* W2b  = (bf16_t*)alloc((size_t)kDFF * kDFF * 2);
  bf16_t* W3b  = (bf16_t*)alloc((size_t)kDV * kDFF * 2);
  float*  Qpf  = (float*)alloc(nTok * 4);
  bf16_t* Qpb  = (bf16_t*)alloc(nTok * 2);
  bf16_t* Kpb  = (bf16_t*)alloc(nTok * 2);
  bf16_t* Vpb  = (bf16_t*)alloc(nTok * 2);
  float*  Of   = (float*)alloc(nTok * 4);
  float*  Xf   = (float*)alloc(nTok * 4);
  bf16_t* Xb   = (bf16_t*)alloc(nTok * 2);
  bf16_t* h1b  = (bf16_t*)alloc(nFF * 2);
  bf16_t* h2b  = (bf16_t*)alloc(nFF * 2);
  float*  Offf = (float*)alloc(nTok * 4);

  auto cast = [&](const float* s, bf16_t* d, size_t n) {
    int blocks = (int)((n + 1023) / 1024);
    cast_f32_bf16<<<blocks, 256, 0, stream>>>(s, d, (int)n);
  };
  cast(Q,  Qbf, nTok);
  cast(K,  Kbf, nTok);
  cast(Wq, Wqb, (size_t)kDV * kDQ);
  cast(Wk, Wkb, (size_t)kDV * kDQ);
  cast(Wv, Wvb, (size_t)kDV * kDQ);
  cast(W1, W1b, (size_t)kDFF * kDV);
  cast(W2, W2b, (size_t)kDFF * kDFF);
  cast(W3, W3b, (size_t)kDV * kDFF);

  const dim3 blk(256);
  const dim3 gProj(kDV / 64, kRows / 128);     // (8, 64)
  const dim3 gFF1(kDFF / 64, kRows / 128);     // (32, 64)
  const dim3 gFF3(kDV / 64, kRows / 128);      // (8, 64)

  // projections
  gemm_bias_act<false, true,  true ><<<gProj, blk, 0, stream>>>(Qbf, Wqb, bq, Qpf, Qpb, kDV, kDQ);
  gemm_bias_act<false, false, true ><<<gProj, blk, 0, stream>>>(Kbf, Wkb, bk, nullptr, Kpb, kDV, kDQ);
  gemm_bias_act<false, false, true ><<<gProj, blk, 0, stream>>>(Kbf, Wvb, bv, nullptr, Vpb, kDV, kDQ);

  // attention
  attention_kernel<<<dim3(kN / 64, kH, kB), dim3(128), 0, stream>>>(Qpb, Kpb, Vpb, Of);

  // LN0: X = LN(O + Qp)
  add_layernorm<true><<<kRows, blk, 0, stream>>>(Of, Qpf, g0, beta0, Xf, Xb);

  // FFN
  gemm_bias_act<true,  false, true ><<<gFF1, blk, 0, stream>>>(Xb,  W1b, b1, nullptr, h1b, kDFF, kDV);
  gemm_bias_act<true,  false, true ><<<gFF1, blk, 0, stream>>>(h1b, W2b, b2, nullptr, h2b, kDFF, kDFF);
  gemm_bias_act<false, true,  false><<<gFF3, blk, 0, stream>>>(h2b, W3b, b3, Offf, nullptr, kDV, kDFF);

  // LN1: out = LN(X + Off)
  add_layernorm<false><<<kRows, blk, 0, stream>>>(Offf, Xf, g1, beta1, out, nullptr);
}
